// LSTM_24816321036344
// MI455X (gfx1250) — matrix operand order repaired and checked by reading, not compiled, not run on hardware
//
#include <hip/hip_runtime.h>

// ---------------------------------------------------------------------------
// Problem constants (from reference): B=64, T=2048, I=H=128, L=3, OUT=10
// ---------------------------------------------------------------------------
static constexpr int Bsz = 64;
static constexpr int T   = 2048;
static constexpr int H   = 128;
static constexpr int G   = 4 * H;        // 512 gate columns
static constexpr int K2  = 2 * H;        // 256 fused K (x ++ h)
static constexpr int BPB = 16;           // batch rows per block (M tile)
static constexpr int THREADS = 1024;     // 32 waves (wave32): 1 N-tile/wave

// ---------------------------------------------------------------------------
// WMMA vector types (gfx1250, wave32)
// ---------------------------------------------------------------------------
typedef __attribute__((ext_vector_type(16))) __bf16          v16bf;
typedef __attribute__((ext_vector_type(8)))  float           v8f;
typedef __attribute__((ext_vector_type(8)))  unsigned short  u16x8;
typedef __attribute__((ext_vector_type(16))) unsigned short  u16x16;

union BfFrag {
    u16x16 u;
    v16bf  b;
    unsigned short s[16];
    struct { u16x8 lo, hi; } h;
};

__device__ __forceinline__ unsigned short f2bf(float f) {
    unsigned int u = __float_as_uint(f);
    unsigned int r = (u + 0x7FFFu + ((u >> 16) & 1u)) >> 16;   // RNE
    return (unsigned short)r;
}
__device__ __forceinline__ float bf2f(unsigned short h) {
    return __uint_as_float(((unsigned int)h) << 16);
}
// Straight-line activations: v_exp_f32 + v_rcp_f32, no ocml branches.
__device__ __forceinline__ float fast_sigmoid(float x) {
    return __builtin_amdgcn_rcpf(1.0f + __expf(-x));
}
__device__ __forceinline__ float fast_tanh(float x) {
    return 1.0f - 2.0f * __builtin_amdgcn_rcpf(1.0f + __expf(2.0f * x));
}

// ---------------------------------------------------------------------------
// Persistent LSTM layer kernel: grid = 4 blocks (16 batch rows each),
// 1024 threads = 32 waves. Wave w owns gate columns [w*16, w*16+16).
// The fused weight operand [W_ih ; W_hh]^T for those columns lives in
// REGISTERS (8 bf16 fragments = 64 VGPRs) for the entire 2048-step loop;
// only the 16x256 activation tile A = [x_t | h_{t-1}] moves through LDS.
// Per timestep: all 8 A fragments are loaded up-front (one dscnt wait),
// then two independent 4-deep WMMA chains accumulate the gate tile.
// FIRST=true  -> input is fp32 (original x)
// FIRST=false -> input is bf16 (previous layer output in workspace)
// ---------------------------------------------------------------------------
template <bool FIRST>
__global__ __launch_bounds__(THREADS, 1)
void lstm_layer_kernel(const void*  __restrict__ in_v,     // x or bf16 buffer
                       const float* __restrict__ wih,      // [512,128]
                       const float* __restrict__ whh,      // [512,128]
                       const float* __restrict__ bih,      // [512]
                       const float* __restrict__ bhh,      // [512]
                       unsigned short* __restrict__ out_bf)// [B,T,H] bf16
{
    __shared__ unsigned short a_bf[BPB * K2];   // [16][256]: x | h   (8 KB)
    __shared__ float          gates[BPB * G];   // [16][512]         (32 KB)

    const int tid   = threadIdx.x;
    const int lane  = tid & 31;
    const int wave  = tid >> 5;        // 0..31  -> N-tile index
    const int half  = lane >> 4;       // 0: lanes 0-15, 1: lanes 16-31
    const int lrow  = lane & 15;
    const int bbase = blockIdx.x * BPB;

    const float*          in_f  = (const float*)in_v;
    const unsigned short* in_bf = (const unsigned short*)in_v;

    // ---- per-wave stationary B operand: column `col` of the fused weight
    //      matrix, 8 k-block fragments, converted fp32 -> bf16 in registers.
    //      B frag k order used here (same as A): lane = column, lanes 0-15 hold
    //      K 0..15 of the k-block, lanes 16-31 hold K 16..31, contiguous.
    const int col = wave * 16 + lrow;          // 0..511
    BfFrag bw[8];
    #pragma unroll
    for (int kb = 0; kb < 8; ++kb) {
        const int kf = kb * 32 + half * 16;    // fused K base (0..240)
        const float* src = (kf < H) ? &wih[col * H + kf]
                                    : &whh[col * H + (kf - H)];
        #pragma unroll
        for (int e = 0; e < 16; ++e) bw[kb].s[e] = f2bf(src[e]);
    }
    const float bias_c = bih[col] + bhh[col];  // per-lane gate bias

    // ---- elementwise / staging mapping: 2 elements per thread -------------
    const int em = tid >> 6;                   // batch row 0..15
    const int jb = (tid & 63) * 2;             // col base 0..126
    float creg[2] = {0.0f, 0.0f};              // cell state, register-resident

    // zero initial h in LDS
    a_bf[em * K2 + H + jb]     = 0;
    a_bf[em * K2 + H + jb + 1] = 0;

    for (int t = 0; t < T; ++t) {
        // ---- stage x_t tile into A (cols 0..127) as bf16 ------------------
        {
            const size_t rowoff = ((size_t)(bbase + em) * T + t) * H + jb;
            if (FIRST) {
                a_bf[em * K2 + jb]     = f2bf(in_f[rowoff]);
                a_bf[em * K2 + jb + 1] = f2bf(in_f[rowoff + 1]);
                if (t + 1 < T) __builtin_prefetch(&in_f[rowoff + H], 0, 3);
            } else {
                a_bf[em * K2 + jb]     = in_bf[rowoff];
                a_bf[em * K2 + jb + 1] = in_bf[rowoff + 1];
                if (t + 1 < T) __builtin_prefetch(&in_bf[rowoff + H], 0, 3);
            }
        }
        __syncthreads();   // x staged + previous h written before GEMM

        // ---- g(:,col16) = A(16x256) @ Wtile(256x16) + bias ----------------
        {
            // Load ALL A fragments first: 16 back-to-back ds_load_b128,
            // a single dscnt wait, then the WMMA chains.
            BfFrag af[8];
            #pragma unroll
            for (int kb = 0; kb < 8; ++kb) {
                const unsigned short* ap =
                    &a_bf[lrow * K2 + kb * 32 + half * 16];
                af[kb].h.lo = *(const u16x8*)(ap);
                af[kb].h.hi = *(const u16x8*)(ap + 8);
            }
            // Two independent accumulation chains (even / odd k-blocks).
            v8f acc0, acc1;
            #pragma unroll
            for (int r = 0; r < 8; ++r) { acc0[r] = bias_c; acc1[r] = 0.0f; }
            #pragma unroll
            for (int kb = 0; kb < 8; kb += 2) {
                acc0 = __builtin_amdgcn_wmma_f32_16x16x32_bf16(
                    false, af[kb].b, false, bw[kb].b, (short)0, acc0,
                    false, false);
                acc1 = __builtin_amdgcn_wmma_f32_16x16x32_bf16(
                    false, af[kb + 1].b, false, bw[kb + 1].b, (short)0, acc1,
                    false, false);
            }
            // D layout: VGPR r -> row r + half*8, col = lrow within tile
            #pragma unroll
            for (int r = 0; r < 8; ++r)
                gates[(r + half * 8) * G + col] = acc0[r] + acc1[r];
        }
        __syncthreads();   // gates complete before elementwise

        // ---- LSTM cell update (fp32), c in registers, h -> LDS + global ---
        {
            const float* grow = &gates[em * G];
            #pragma unroll
            for (int e = 0; e < 2; ++e) {
                const int j = jb + e;
                const float gi = fast_sigmoid(grow[j]);
                const float gf = fast_sigmoid(grow[H + j]);
                const float gg = fast_tanh(grow[2 * H + j]);
                const float go = fast_sigmoid(grow[3 * H + j]);
                const float c  = gf * creg[e] + gi * gg;
                creg[e] = c;
                const float h  = go * fast_tanh(c);
                const unsigned short hb = f2bf(h);
                a_bf[em * K2 + H + j] = hb;                      // h for t+1
                out_bf[((size_t)(bbase + em) * T + t) * H + j] = hb;
            }
        }
        // gates is only re-written after the top-of-loop barrier, and the
        // staging region of a_bf is disjoint from the h region: safe.
    }
}

// ---------------------------------------------------------------------------
// Linear head: h_last(64x128) -> 64 -> 32 -> 10   (no nonlinearity)
// One thread per batch row; all weights fp32 from global (tiny).
// ---------------------------------------------------------------------------
__global__ __launch_bounds__(64, 1)
void head_kernel(const unsigned short* __restrict__ hbuf,  // [B,T,H] bf16
                 const float* __restrict__ w1, const float* __restrict__ b1,
                 const float* __restrict__ w2, const float* __restrict__ b2,
                 const float* __restrict__ w3, const float* __restrict__ b3,
                 float* __restrict__ out)                    // [B,10]
{
    const int b = threadIdx.x;
    if (b >= Bsz) return;
    float h[H];
    const unsigned short* hp = hbuf + ((size_t)b * T + (T - 1)) * H;
    #pragma unroll 8
    for (int j = 0; j < H; ++j) h[j] = bf2f(hp[j]);

    float t1[64];
    for (int o = 0; o < 64; ++o) {
        float s = b1[o];
        #pragma unroll 8
        for (int j = 0; j < H; ++j) s = fmaf(h[j], w1[o * H + j], s);
        t1[o] = s;
    }
    float t2[32];
    for (int o = 0; o < 32; ++o) {
        float s = b2[o];
        #pragma unroll 8
        for (int j = 0; j < 64; ++j) s = fmaf(t1[j], w2[o * 64 + j], s);
        t2[o] = s;
    }
    for (int o = 0; o < 10; ++o) {
        float s = b3[o];
        #pragma unroll 8
        for (int j = 0; j < 32; ++j) s = fmaf(t2[j], w3[o * 32 + j], s);
        out[b * 10 + o] = s;
    }
}

// ---------------------------------------------------------------------------
// Host launcher
// in order: x, w_ih[L], w_hh[L], b_ih[L], b_hh[L], w1,b1,w2,b2,w3,b3
// ---------------------------------------------------------------------------
extern "C" void kernel_launch(void* const* d_in, const int* in_sizes, int n_in,
                              void* d_out, int out_size, void* d_ws,
                              size_t ws_size, hipStream_t stream)
{
    (void)in_sizes; (void)n_in; (void)out_size; (void)ws_size;

    const float* x    = (const float*)d_in[0];
    const float* w_ih = (const float*)d_in[1];   // [L,512,128]
    const float* w_hh = (const float*)d_in[2];   // [L,512,128]
    const float* b_ih = (const float*)d_in[3];   // [L,512]
    const float* b_hh = (const float*)d_in[4];   // [L,512]
    const float* w1   = (const float*)d_in[5];
    const float* b1   = (const float*)d_in[6];
    const float* w2   = (const float*)d_in[7];
    const float* b2   = (const float*)d_in[8];
    const float* w3   = (const float*)d_in[9];
    const float* b3   = (const float*)d_in[10];
    float* out = (float*)d_out;

    // workspace: two bf16 [B,T,H] ping-pong buffers (32 MB each)
    const size_t BUF_ELEMS = (size_t)Bsz * T * H;
    unsigned short* buf0 = (unsigned short*)d_ws;
    unsigned short* buf1 = buf0 + BUF_ELEMS;

    const dim3 grid(Bsz / BPB), block(THREADS);

    // layer 0: fp32 x -> buf0
    lstm_layer_kernel<true><<<grid, block, 0, stream>>>(
        (const void*)x,
        w_ih + 0 * (size_t)G * H, w_hh + 0 * (size_t)G * H,
        b_ih + 0 * G, b_hh + 0 * G, buf0);
    // layer 1: buf0 -> buf1
    lstm_layer_kernel<false><<<grid, block, 0, stream>>>(
        (const void*)buf0,
        w_ih + 1 * (size_t)G * H, w_hh + 1 * (size_t)G * H,
        b_ih + 1 * G, b_hh + 1 * G, buf1);
    // layer 2: buf1 -> buf0
    lstm_layer_kernel<false><<<grid, block, 0, stream>>>(
        (const void*)buf1,
        w_ih + 2 * (size_t)G * H, w_hh + 2 * (size_t)G * H,
        b_ih + 2 * G, b_hh + 2 * G, buf0);

    head_kernel<<<dim3(1), dim3(64), 0, stream>>>(
        buf0, w1, b1, w2, b2, w3, b3, out);
}
